// Pointnet2_6_35828617183508
// MI455X (gfx1250) — matrix operand order repaired and checked by reading, not compiled, not run on hardware
//
#include <hip/hip_runtime.h>
#include <hip/hip_bf16.h>
#include <math.h>

typedef __attribute__((ext_vector_type(16))) _Float16 v16h;
typedef __attribute__((ext_vector_type(8)))  _Float16 v8h;
typedef __attribute__((ext_vector_type(8)))  float    v8f;

static constexpr int   BATCH    = 2;
static constexpr int   NPTS0    = 16384;
static constexpr float BN_EPS_C = 1e-5f;

static __host__ __device__ inline int rup32(int x) { return (x + 31) & ~31; }
static __host__ __device__ inline int rup16(int x) { return (x + 15) & ~15; }

// ---------------------------------------------------------------------------
// (B,3,N) -> (B,N,3)
__global__ void transpose_xyz_kernel(const float* __restrict__ in,
                                     float* __restrict__ out, int N) {
  int tid = blockIdx.x * blockDim.x + threadIdx.x;
  int total = BATCH * N;
  if (tid >= total) return;
  int b = tid / N, n = tid - b * N;
#pragma unroll
  for (int c = 0; c < 3; ++c)
    out[(size_t)tid * 3 + c] = in[((size_t)b * 3 + c) * N + n];
}

// ---------------------------------------------------------------------------
// Farthest point sampling: one block per batch. The level's xyz is staged
// into LDS with gfx1250 async-to-LDS loads (ASYNCcnt), and the running
// min-distance array lives in LDS as well, so the npoint-long serial
// recurrence runs entirely out of LDS.
extern __shared__ float fps_smem[];
__global__ void fps_kernel(const float* __restrict__ xyz, int N, int npoint,
                           int* __restrict__ out_idx) {
  float* s_xyz  = fps_smem;           // N*3 floats
  float* s_dist = fps_smem + N * 3;   // N floats
  __shared__ float s_val[256];
  __shared__ int   s_idx[256];
  int b = blockIdx.x, t = threadIdx.x;
  const float* p = xyz + (size_t)b * N * 3;

  // async stage: 16B per lane per issue, wave waits its own ASYNCcnt,
  // then the workgroup barrier makes all staged data visible.
  {
    int nbytes = N * 12;
    unsigned lds0 = (unsigned)(size_t)s_xyz;
    unsigned long long g0 = (unsigned long long)(size_t)p;
    for (int o = t * 16; o < nbytes; o += 256 * 16) {
      unsigned ldsa = lds0 + (unsigned)o;
      unsigned long long ga = g0 + (unsigned long long)o;
      asm volatile("global_load_async_to_lds_b128 %0, %1, off"
                   :: "v"(ldsa), "v"(ga) : "memory");
    }
    asm volatile("s_wait_asynccnt 0" ::: "memory");
  }
  for (int i = t; i < N; i += 256) s_dist[i] = 1e10f;
  __syncthreads();

  int far = 0;
  for (int it = 0; it < npoint; ++it) {
    if (t == 0) out_idx[b * npoint + it] = far;
    float cx = s_xyz[far*3+0], cy = s_xyz[far*3+1], cz = s_xyz[far*3+2];
    float best = -1.0f; int bi = 0x7fffffff;
    for (int i = t; i < N; i += 256) {
      float dx = s_xyz[i*3+0]-cx, dy = s_xyz[i*3+1]-cy, dz = s_xyz[i*3+2]-cz;
      float d = dx*dx + dy*dy + dz*dz;
      float m = fminf(s_dist[i], d);
      s_dist[i] = m;
      if (m > best) { best = m; bi = i; }
    }
    s_val[t] = best; s_idx[t] = bi;
    __syncthreads();
    for (int s = 128; s > 0; s >>= 1) {
      if (t < s) {
        float ov = s_val[t+s]; int oi = s_idx[t+s];
        if (ov > s_val[t] || (ov == s_val[t] && oi < s_idx[t])) {
          s_val[t] = ov; s_idx[t] = oi;
        }
      }
      __syncthreads();
    }
    far = s_idx[0];
    __syncthreads();
  }
}

// ---------------------------------------------------------------------------
__global__ void gather_xyz_kernel(const float* __restrict__ xyz,
                                  const int* __restrict__ idx,
                                  float* __restrict__ out, int N, int S) {
  int tid = blockIdx.x * blockDim.x + threadIdx.x;
  if (tid >= BATCH * S) return;
  int b = tid / S;
  int g = idx[tid];
#pragma unroll
  for (int c = 0; c < 3; ++c)
    out[(size_t)tid * 3 + c] = xyz[((size_t)b * N + g) * 3 + c];
}

// ---------------------------------------------------------------------------
// Ball query: one wave per centroid; first K in-radius indices (ascending),
// padded with the first hit -- identical to reference sort/truncate/pad.
__global__ void query_ball_kernel(const float* __restrict__ xyz,
                                  const float* __restrict__ nxyz,
                                  int* __restrict__ gidx,
                                  int N, int S, int K, float r2) {
  int w = blockIdx.x * (blockDim.x >> 5) + (threadIdx.x >> 5);
  int lane = threadIdx.x & 31;
  if (w >= BATCH * S) return;
  int b = w / S;
  const float* p = xyz + (size_t)b * N * 3;
  float cx = nxyz[w*3+0], cy = nxyz[w*3+1], cz = nxyz[w*3+2];
  int* out = gidx + (size_t)w * K;
  int cnt = 0, firstIdx = 0;
  for (int base = 0; base < N && cnt < K; base += 32) {
    int i = base + lane;
    bool inr = false;
    if (i < N) {
      float dx = p[i*3+0]-cx, dy = p[i*3+1]-cy, dz = p[i*3+2]-cz;
      inr = (dx*dx + dy*dy + dz*dz) <= r2;
    }
    unsigned mask = (unsigned)__ballot(inr);
    if (cnt == 0 && mask) firstIdx = base + __ffs(mask) - 1;
    int pos = cnt + __popc(mask & ((1u << lane) - 1u));
    if (inr && pos < K) out[pos] = i;
    cnt += __popc(mask);
  }
  if (cnt > K) cnt = K;
  for (int j = cnt + lane; j < K; j += 32) out[j] = firstIdx;
}

// ---------------------------------------------------------------------------
// feats f16 (B*S*K, Kp) = [ points[gidx] , xyz[gidx]-new_xyz , 0-pad ]
__global__ void group_feats_kernel(const float* __restrict__ xyz,
                                   const float* __restrict__ nxyz,
                                   const float* __restrict__ pts,
                                   const int* __restrict__ gidx,
                                   _Float16* __restrict__ feats,
                                   int N, int S, int K, int Cp, int Kp) {
  int tid = blockIdx.x * blockDim.x + threadIdx.x;
  int total = BATCH * S * K;
  if (tid >= total) return;
  int ws = tid / K;
  int b = ws / S;
  int g = gidx[tid];
  _Float16* out = feats + (size_t)tid * Kp;
  const float* src = pts + ((size_t)b * N + g) * Cp;
  for (int c = 0; c < Cp; ++c) out[c] = (_Float16)src[c];
  const float* px = xyz + ((size_t)b * N + g) * 3;
  const float* cx = nxyz + (size_t)ws * 3;
  out[Cp+0] = (_Float16)(px[0] - cx[0]);
  out[Cp+1] = (_Float16)(px[1] - cx[1]);
  out[Cp+2] = (_Float16)(px[2] - cx[2]);
  for (int c = Cp + 3; c < Kp; ++c) out[c] = (_Float16)0.f;
}

// ---------------------------------------------------------------------------
// Pack f32 weights (N,K) -> f16 (N,Kp), zero-padded K.
__global__ void convert_w_kernel(const float* __restrict__ W,
                                 _Float16* __restrict__ Wh,
                                 int N, int K, int Kp) {
  int tid = blockIdx.x * blockDim.x + threadIdx.x;
  int total = N * Kp;
  if (tid >= total) return;
  int n = tid / Kp, k = tid - n * Kp;
  Wh[tid] = (k < K) ? (_Float16)W[(size_t)n * K + k] : (_Float16)0.f;
}

// ---------------------------------------------------------------------------
// WMMA GEMM, NT sub-tiles of 16 columns per wave (A fragment reused NT times).
// X, Wh are f16 with K padded to 32 and zero-filled: branch-free inner loop.
// mode 0: folded BN + ReLU; mode 1: bias + sigmoid.
// outHalf: f16 store, ldY = rup32(N), pad cols [N,ldY) zeroed; else f32, ldY=N.
template <int NT>
__global__ __launch_bounds__(128)
void gemm_wmma_kernel(const _Float16* __restrict__ X, const _Float16* __restrict__ Wh,
                      const float* __restrict__ bias,
                      const float* __restrict__ gamma, const float* __restrict__ beta,
                      const float* __restrict__ rmean, const float* __restrict__ rvar,
                      void* __restrict__ Yout,
                      int M, int Kp, int N, int Ncover, int ldY,
                      int mode, int outHalf) {
  const int lane = threadIdx.x & 31;
  const int wid  = threadIdx.x >> 5;
  const int groupsN = Ncover / (16 * NT);
  int tile = blockIdx.x * 4 + wid;
  if (tile >= (M >> 4) * groupsN) return;       // wave-uniform -> EXEC all-1s
  const int tm = tile / groupsN;
  const int tg = tile - tm * groupsN;

  const int hsel = lane >> 4;
  const int l15  = lane & 15;
  const int row  = (tm << 4) + l15;             // always < M
  const int colBase = tg * (16 * NT);
  const _Float16* __restrict__ xrow = X + (size_t)row * Kp;
  const _Float16* wrow[NT];
  int col[NT];
#pragma unroll
  for (int i = 0; i < NT; ++i) {
    col[i] = colBase + 16 * i + l15;
    wrow[i] = Wh + (size_t)((col[i] < N) ? col[i] : 0) * Kp;  // clamp
  }
  const int kbA = hsel << 4;                    // A: halves kb..kb+15
  const int kbB = hsel << 4;                    // B: halves kb..kb+15

  v8f acc[NT];
#pragma unroll
  for (int i = 0; i < NT; ++i) acc[i] = (v8f){0.f,0.f,0.f,0.f,0.f,0.f,0.f,0.f};

  for (int k0 = 0; k0 < Kp; k0 += 32) {
    v8h alo = *(const v8h*)(xrow + k0 + kbA);
    v8h ahi = *(const v8h*)(xrow + k0 + 8 + kbA);
    v16h a = __builtin_shufflevector(alo, ahi,
                                     0,1,2,3,4,5,6,7,8,9,10,11,12,13,14,15);
#pragma unroll
    for (int i = 0; i < NT; ++i) {
      v16h bfr = *(const v16h*)(wrow[i] + k0 + kbB);
      acc[i] = __builtin_amdgcn_wmma_f32_16x16x32_f16(
          false, a, false, bfr, (short)0, acc[i], false, false);
    }
  }

#pragma unroll
  for (int i = 0; i < NT; ++i) {
    const bool ok = (col[i] < N);
    float s = 1.f, t = 0.f;
    if (ok) {
      if (mode == 0) {
        float g  = gamma[col[i]], bt = beta[col[i]];
        float mu = rmean[col[i]], vr = rvar[col[i]], bb = bias[col[i]];
        s = g * rsqrtf(vr + BN_EPS_C);
        t = s * (bb - mu) + bt;
      } else {
        t = bias[col[i]];
      }
    }
    if (outHalf) {
      _Float16* Yh = (_Float16*)Yout;
#pragma unroll
      for (int v = 0; v < 8; ++v) {
        int r = (tm << 4) + v + (hsel << 3);
        float y = s * acc[i][v] + t;
        y = (mode == 0) ? fmaxf(y, 0.f) : (1.f / (1.f + expf(-y)));
        if (col[i] < ldY)
          Yh[(size_t)r * ldY + col[i]] = ok ? (_Float16)y : (_Float16)0.f;
      }
    } else if (ok) {
      float* Yf = (float*)Yout;
#pragma unroll
      for (int v = 0; v < 8; ++v) {
        int r = (tm << 4) + v + (hsel << 3);
        float y = s * acc[i][v] + t;
        y = (mode == 0) ? fmaxf(y, 0.f) : (1.f / (1.f + expf(-y)));
        Yf[(size_t)r * ldY + col[i]] = y;
      }
    }
  }
}

// ---------------------------------------------------------------------------
// max over K neighbors of f16 activations (row stride ld), scatter f32 slice.
__global__ void maxpool_kernel(const _Float16* __restrict__ act, float* __restrict__ out,
                               int S, int K, int C, int ld, int Ctot, int coff) {
  int tid = blockIdx.x * blockDim.x + threadIdx.x;
  int total = BATCH * S * C;
  if (tid >= total) return;
  int c = tid % C;
  int ws = tid / C;
  const _Float16* a = act + (size_t)ws * K * ld + c;
  float m = (float)a[0];
  for (int k = 1; k < K; ++k) m = fmaxf(m, (float)a[(size_t)k * ld]);
  out[(size_t)ws * Ctot + coff + c] = m;
}

// ---------------------------------------------------------------------------
__global__ void three_nn_kernel(const float* __restrict__ xyz1,
                                const float* __restrict__ xyz2,
                                int* __restrict__ idx3, float* __restrict__ w3,
                                int N1, int N2) {
  int tid = blockIdx.x * blockDim.x + threadIdx.x;
  if (tid >= BATCH * N1) return;
  int b = tid / N1;
  const float* p1 = xyz1 + (size_t)tid * 3;
  const float* p2 = xyz2 + (size_t)b * N2 * 3;
  float x = p1[0], y = p1[1], z = p1[2];
  float d0 = 1e30f, d1 = 1e30f, d2 = 1e30f;
  int   i0 = 0,     i1 = 0,     i2 = 0;
  for (int j = 0; j < N2; ++j) {
    float dx = x - p2[j*3+0], dy = y - p2[j*3+1], dz = z - p2[j*3+2];
    float d = dx*dx + dy*dy + dz*dz;
    if (d < d0)      { d2=d1; i2=i1; d1=d0; i1=i0; d0=d; i0=j; }
    else if (d < d1) { d2=d1; i2=i1; d1=d;  i1=j; }
    else if (d < d2) { d2=d;  i2=j; }
  }
  float r0 = 1.f/(d0+1e-8f), r1 = 1.f/(d1+1e-8f), r2 = 1.f/(d2+1e-8f);
  float s = r0 + r1 + r2;
  size_t o = (size_t)tid * 3;
  idx3[o+0] = i0; idx3[o+1] = i1; idx3[o+2] = i2;
  w3[o+0] = r0/s; w3[o+1] = r1/s; w3[o+2] = r2/s;
}

// weighted 3-NN interpolation -> f16 slice [coff, coff+C2) of row stride ld
__global__ void interp_kernel(const float* __restrict__ pts2,
                              const int* __restrict__ idx3,
                              const float* __restrict__ w3,
                              _Float16* __restrict__ out,
                              int N1, int N2, int C2, int ld, int coff) {
  int tid = blockIdx.x * blockDim.x + threadIdx.x;
  int total = BATCH * N1 * C2;
  if (tid >= total) return;
  int c = tid % C2;
  int bn = tid / C2;
  int b = bn / N1;
  const int*   id = idx3 + (size_t)bn * 3;
  const float* w  = w3   + (size_t)bn * 3;
  const float* p  = pts2 + (size_t)b * N2 * C2;
  float v = w[0] * p[(size_t)id[0] * C2 + c]
          + w[1] * p[(size_t)id[1] * C2 + c]
          + w[2] * p[(size_t)id[2] * C2 + c];
  out[(size_t)bn * ld + coff + c] = (_Float16)v;
}

__global__ void copy_channels_kernel(const float* __restrict__ src,
                                     _Float16* __restrict__ dst,
                                     int BN, int Csrc, int ld, int coff) {
  int tid = blockIdx.x * blockDim.x + threadIdx.x;
  int total = BN * Csrc;
  if (tid >= total) return;
  int c = tid % Csrc;
  int bn = tid / Csrc;
  dst[(size_t)bn * ld + coff + c] = (_Float16)src[(size_t)bn * Csrc + c];
}

__global__ void fill_pad_h_kernel(_Float16* __restrict__ dst, int rows,
                                  int start, int ld) {
  int pad = ld - start;
  int tid = blockIdx.x * blockDim.x + threadIdx.x;
  int total = rows * pad;
  if (tid >= total) return;
  int c = tid % pad;
  int bn = tid / pad;
  dst[(size_t)bn * ld + start + c] = (_Float16)0.f;
}

// ---------------------------------------------------------------------------
struct Layer { const float *W, *b, *g, *bt, *rm, *rv; };

extern "C" void kernel_launch(void* const* d_in, const int* in_sizes, int n_in,
                              void* d_out, int out_size, void* d_ws, size_t ws_size,
                              hipStream_t stream) {
  (void)in_sizes; (void)n_in; (void)out_size; (void)ws_size;

  // ---- parse inputs (setup_inputs insertion order) ----
  int pi = 0;
  auto F = [&](int i){ return (const float*)d_in[i]; };
  const float* xyz_in = F(pi++);
  Layer sa[6][2][3];
  for (int l = 0; l < 6; ++l)
    for (int br = 0; br < 2; ++br)
      for (int ly = 0; ly < 3; ++ly) {
        sa[l][br][ly] = { F(pi), F(pi+1), F(pi+2), F(pi+3), F(pi+4), F(pi+5) };
        pi += 6;
      }
  static const int fpn[6] = {2,2,2,2,2,3};
  Layer fpl[6][3] = {};
  for (int j = 0; j < 6; ++j)
    for (int ly = 0; ly < fpn[j]; ++ly) {
      fpl[j][ly] = { F(pi), F(pi+1), F(pi+2), F(pi+3), F(pi+4), F(pi+5) };
      pi += 6;
    }
  Layer h1 = { F(pi), F(pi+1), F(pi+2), F(pi+3), F(pi+4), F(pi+5) }; pi += 6;
  Layer h2 = { F(pi), F(pi+1), nullptr, nullptr, nullptr, nullptr }; pi += 2;

  // ---- configs ----
  static const int   nL[7]  = {16384,1024,512,256,64,16,8};
  static const int   cLv[7] = {3,48,96,256,512,1024,2048};
  static const float rad[6][2] = {{0.025f,0.05f},{0.05f,0.1f},{0.1f,0.2f},
                                  {0.2f,0.4f},{0.4f,0.8f},{0.8f,1.0f}};
  static const int   nsK[6][2] = {{16,32},{16,32},{16,32},{16,32},{16,32},{16,16}};
  static const int   mlp[6][2][3] = {
    {{8,8,16},{16,16,32}}, {{16,16,32},{32,32,64}}, {{64,64,128},{64,96,128}},
    {{128,196,256},{128,192,256}}, {{256,256,512},{256,384,512}},
    {{512,512,1024},{512,768,1024}}};
  static const int fpin[6]    = {3072,1024,512,352,176,64};
  static const int fpch[6][3] = {{512,512,0},{256,256,0},{256,256,0},
                                 {256,128,0},{128,64,0},{64,64,64}};

  // ---- workspace carve (bump allocator, 256B aligned) ----
  size_t off = 0;
  auto alloc = [&](size_t bytes) -> char* {
    char* r = (char*)d_ws + off;
    off += (bytes + 255) & ~(size_t)255;
    return r;
  };
  float* xyzL[7]; float* ptsL[7];
  for (int l = 0; l < 7; ++l)
    xyzL[l] = (float*)alloc((size_t)BATCH * nL[l] * 3 * sizeof(float));
  ptsL[0] = xyzL[0];
  for (int l = 1; l < 7; ++l)
    ptsL[l] = (float*)alloc((size_t)BATCH * nL[l] * cLv[l] * sizeof(float));
  int*      fpsIdx = (int*)     alloc((size_t)BATCH * 1024 * sizeof(int));
  int*      gidx   = (int*)     alloc((size_t)BATCH * 1024 * 32 * sizeof(int));
  _Float16* gfH    = (_Float16*)alloc((size_t)2228224 * sizeof(_Float16));
  _Float16* a0H    = (_Float16*)alloc((size_t)2228224 * sizeof(_Float16));
  _Float16* a1H    = (_Float16*)alloc((size_t)2228224 * sizeof(_Float16));
  int*      idx3   = (int*)     alloc((size_t)BATCH * NPTS0 * 3 * sizeof(int));
  float*    w3     = (float*)   alloc((size_t)BATCH * NPTS0 * 3 * sizeof(float));
  _Float16* fpcatH[6]; float* fpoutF[5]; _Float16* fpoutH;
  for (int j = 0; j < 6; ++j) {
    int lvl = 5 - j;
    fpcatH[j] = (_Float16*)alloc((size_t)BATCH * nL[lvl] * rup32(fpin[j]) * sizeof(_Float16));
    if (j < 5)
      fpoutF[j] = (float*)alloc((size_t)BATCH * nL[lvl] * fpch[j][fpn[j]-1] * sizeof(float));
  }
  fpoutH = (_Float16*)alloc((size_t)BATCH * NPTS0 * 64 * sizeof(_Float16));

  auto cdiv = [](int a, int b){ return (a + b - 1) / b; };

  // ---- pack all weights to f16 (padded K) ----
  auto prepW = [&](const Layer& L, int N, int K) -> _Float16* {
    int Kp = rup32(K);
    _Float16* Wh = (_Float16*)alloc((size_t)N * Kp * sizeof(_Float16));
    convert_w_kernel<<<cdiv(N*Kp,256),256,0,stream>>>(L.W, Wh, N, K, Kp);
    return Wh;
  };
  _Float16* saW[6][2][3];
  for (int l = 0; l < 6; ++l)
    for (int br = 0; br < 2; ++br)
      for (int ly = 0; ly < 3; ++ly) {
        int K = (ly == 0) ? (cLv[l] + 3) : mlp[l][br][ly-1];
        saW[l][br][ly] = prepW(sa[l][br][ly], mlp[l][br][ly], K);
      }
  _Float16* fpW[6][3] = {};
  for (int j = 0; j < 6; ++j)
    for (int ly = 0; ly < fpn[j]; ++ly) {
      int K = (ly == 0) ? fpin[j] : fpch[j][ly-1];
      fpW[j][ly] = prepW(fpl[j][ly], fpch[j][ly], K);
    }
  _Float16* h1W = prepW(h1, 64, 64);
  _Float16* h2W = prepW(h2, 3, 64);

  auto gemm = [&](const _Float16* X, const _Float16* Wh, const Layer& L,
                  void* Y, int M, int K, int N, int mode, int outHalf) {
    int Kp  = rup32(K);
    int ldY = outHalf ? rup32(N) : N;
    int NT  = (rup16(N) >= 64) ? 4 : 1;
    int unit = 16 * NT;
    int base = outHalf ? ldY : N;
    int Ncover = (base + unit - 1) / unit * unit;
    int tiles = (M >> 4) * (Ncover / unit);
    if (NT == 4)
      gemm_wmma_kernel<4><<<cdiv(tiles,4),128,0,stream>>>(
          X, Wh, L.b, L.g, L.bt, L.rm, L.rv, Y, M, Kp, N, Ncover, ldY, mode, outHalf);
    else
      gemm_wmma_kernel<1><<<cdiv(tiles,4),128,0,stream>>>(
          X, Wh, L.b, L.g, L.bt, L.rm, L.rv, Y, M, Kp, N, Ncover, ldY, mode, outHalf);
  };

  // ---- input transpose ----
  transpose_xyz_kernel<<<cdiv(BATCH*NPTS0,256),256,0,stream>>>(xyz_in, xyzL[0], NPTS0);

  // ---- set abstraction (MSG) ----
  for (int l = 0; l < 6; ++l) {
    int N = nL[l], S = nL[l+1];
    fps_kernel<<<BATCH,256,(size_t)N*16,stream>>>(xyzL[l], N, S, fpsIdx);
    gather_xyz_kernel<<<cdiv(BATCH*S,256),256,0,stream>>>(xyzL[l], fpsIdx, xyzL[l+1], N, S);
    int coff = 0;
    for (int br = 0; br < 2; ++br) {
      int K = nsK[l][br]; float r = rad[l][br];
      query_ball_kernel<<<cdiv(BATCH*S,8),256,0,stream>>>(
          xyzL[l], xyzL[l+1], gidx, N, S, K, r*r);
      int Cp = cLv[l];
      int cin = Cp + 3, Kp0 = rup32(cin);
      group_feats_kernel<<<cdiv(BATCH*S*K,256),256,0,stream>>>(
          xyzL[l], xyzL[l+1], ptsL[l], gidx, gfH, N, S, K, Cp, Kp0);
      int M = BATCH * S * K;
      const _Float16* cur = gfH; int kc = cin;
      _Float16* bufs[2] = {a0H, a1H};
      for (int ly = 0; ly < 3; ++ly) {
        _Float16* dst = bufs[ly & 1];
        gemm(cur, saW[l][br][ly], sa[l][br][ly], dst, M, kc, mlp[l][br][ly], 0, 1);
        cur = dst; kc = mlp[l][br][ly];
      }
      maxpool_kernel<<<cdiv(BATCH*S*kc,256),256,0,stream>>>(
          cur, ptsL[l+1], S, K, kc, rup32(kc), cLv[l+1], coff);
      coff += kc;
    }
  }

  // ---- feature propagation ----
  const float* feats = ptsL[6]; int cf = cLv[6];
  for (int j = 0; j < 6; ++j) {
    int lvl = 5 - j, N1 = nL[lvl], N2 = nL[lvl+1];
    three_nn_kernel<<<cdiv(BATCH*N1,256),256,0,stream>>>(
        xyzL[lvl], xyzL[lvl+1], idx3, w3, N1, N2);
    int c1 = (lvl >= 1) ? cLv[lvl] : 0;
    int Ctot = c1 + cf;                 // == fpin[j]
    int ld = rup32(Ctot);
    if (c1 > 0)
      copy_channels_kernel<<<cdiv(BATCH*N1*c1,256),256,0,stream>>>(
          ptsL[lvl], fpcatH[j], BATCH*N1, c1, ld, 0);
    interp_kernel<<<cdiv(BATCH*N1*cf,256),256,0,stream>>>(
        feats, idx3, w3, fpcatH[j], N1, N2, cf, ld, c1);
    if (ld > Ctot)
      fill_pad_h_kernel<<<cdiv(BATCH*N1*(ld-Ctot),256),256,0,stream>>>(
          fpcatH[j], BATCH*N1, Ctot, ld);
    const _Float16* cur = fpcatH[j]; int kc = Ctot;
    _Float16* bufs[2] = {a0H, a1H};
    for (int ly = 0; ly < fpn[j]; ++ly) {
      bool last = (ly == fpn[j] - 1);
      if (last && j < 5) {
        gemm(cur, fpW[j][ly], fpl[j][ly], fpoutF[j], BATCH*N1, kc, fpch[j][ly], 0, 0);
      } else {
        _Float16* dst = last ? fpoutH : bufs[ly & 1];
        gemm(cur, fpW[j][ly], fpl[j][ly], dst, BATCH*N1, kc, fpch[j][ly], 0, 1);
        if (!last) cur = dst;
      }
      kc = fpch[j][ly];
    }
    if (j < 5) { feats = fpoutF[j]; cf = kc; }
  }

  // ---- head: conv1 (BN+ReLU, f16 out) then conv2 (bias+sigmoid, f32 out) ----
  gemm(fpoutH, h1W, h1, a0H, BATCH*NPTS0, 64, 64, 0, 1);
  gemm(a0H, h2W, h2, (float*)d_out, BATCH*NPTS0, 64, 3, 1, 0);
}
